// Attention_88141318848873
// MI455X (gfx1250) — matrix operand order repaired and checked by reading, not compiled, not run on hardware
//
#include <hip/hip_runtime.h>

#define DEVINL __device__ __forceinline__

typedef __attribute__((ext_vector_type(16))) __bf16 bf16x16;
typedef __attribute__((ext_vector_type(8)))  __bf16 bf16x8;
typedef __attribute__((ext_vector_type(8)))  float  f32x8;

constexpr int Bc  = 2;
constexpr int Sc  = 2048;
constexpr int Hc  = 2048;
constexpr int NHc = 32;
constexpr int HDc = 64;

constexpr int BROW = 40;   // padded LDS row stride (elements) for B tiles

// ---------------- bf16 helpers (RNE) ----------------
DEVINL unsigned short f2bf_bits(float f) {
  unsigned int u = __float_as_uint(f);
  u += 0x7FFFu + ((u >> 16) & 1u);
  return (unsigned short)(u >> 16);
}
DEVINL __bf16 f2bf(float f) {
  unsigned short b = f2bf_bits(f);
  return __builtin_bit_cast(__bf16, b);
}
DEVINL float bf2f(unsigned short b) {
  return __uint_as_float(((unsigned int)b) << 16);
}

// ---------------- WMMA wrapper ----------------
DEVINL f32x8 wmma_bf16(bf16x16 a, bf16x16 b, f32x8 c) {
  return __builtin_amdgcn_wmma_f32_16x16x32_bf16(
      /*neg_a=*/false, a, /*neg_b=*/false, b,
      /*c_mod=*/(short)0, c, /*reuse_a=*/false, /*reuse_b=*/false);
}

// B fragment (32x16 K-major): lane holds 16 contiguous K elems of one column.
DEVINL bf16x16 bfrag(const unsigned short* p) {
  return *reinterpret_cast<const bf16x16*>(p);
}
// A fragment (16x32): lane holds two 8-elem contiguous K chunks of one row.
DEVINL bf16x16 afrag_bf16(const unsigned short* rowp, int kc0, int kc1) {
  bf16x8 c0 = *reinterpret_cast<const bf16x8*>(rowp + kc0);
  bf16x8 c1 = *reinterpret_cast<const bf16x8*>(rowp + kc1);
  bf16x16 a;
#pragma unroll
  for (int i = 0; i < 8; ++i) { a[i] = c0[i]; a[i + 8] = c1[i]; }
  return a;
}
// A fragment with on-the-fly fp32 -> bf16 conversion.
DEVINL bf16x16 afrag_f32(const float* rowp, int kc0, int kc1) {
  f32x8 f0 = *reinterpret_cast<const f32x8*>(rowp + kc0);
  f32x8 f1 = *reinterpret_cast<const f32x8*>(rowp + kc1);
  bf16x16 a;
#pragma unroll
  for (int i = 0; i < 8; ++i) { a[i] = f2bf(f0[i]); a[i + 8] = f2bf(f1[i]); }
  return a;
}

// ---------------- async copy of one 64(N) x 32(K) bf16 B tile into LDS ----------------
// Each of the 128 threads moves 32 bytes (two async b128 ops, ASYNCcnt += 2/wave).
// INST_OFFSET is added to both the LDS and the global address (ISA 08, §4.4).
DEVINL void async_copy_b_tile(const unsigned short* w16, int n0, int k0,
                              unsigned short* lds_buf, int tid) {
  const int n  = tid >> 1;
  const int kh = tid & 1;
  const unsigned short* src = w16 + (size_t)(n0 + n) * Hc + k0 + 16 * kh;
  const unsigned dst =
      (unsigned)(uintptr_t)(lds_buf + n * BROW + 16 * kh);  // LDS aperture offset
  const unsigned long long ga = (unsigned long long)(uintptr_t)src;
  asm volatile("global_load_async_to_lds_b128 %0, %1, off"
               :: "v"(dst), "v"(ga) : "memory");
  asm volatile("global_load_async_to_lds_b128 %0, %1, off offset:16"
               :: "v"(dst), "v"(ga) : "memory");
}
DEVINL void wait_async_le2() { asm volatile("s_wait_asynccnt 0x2" ::: "memory"); }
DEVINL void wait_async_le0() { asm volatile("s_wait_asynccnt 0x0" ::: "memory"); }

// ---------------- 1) fp32 -> bf16 conversion ----------------
__global__ void f32_to_bf16_kernel(const float* __restrict__ src,
                                   unsigned short* __restrict__ dst, int n) {
  int i = blockIdx.x * blockDim.x + threadIdx.x;
  if (i < n) dst[i] = f2bf_bits(src[i]);
}

// ---------------- 2) QKV GEMM + scatter into attention layouts ----------------
// Block = 4 waves sharing one N=64 macro-column; each wave owns 32 M rows.
// B tiles double-buffered in LDS via async-to-LDS DMA.
//   n in [0,H)    -> Qp[b, h, s, d]   (bf16)
//   n in [H,2H)   -> Kp[b, h, s, d]   (bf16)
//   n in [2H,3H)  -> Vt[b, h, d, s]   (bf16, transposed for B-fragments)
__global__ __launch_bounds__(128) void gemm_qkv_kernel(
    const float* __restrict__ hidden, const unsigned short* __restrict__ w16,
    unsigned short* __restrict__ Qp, unsigned short* __restrict__ Kp,
    unsigned short* __restrict__ Vt) {
  __shared__ unsigned short Bsh[2][64 * BROW];
  const int tid  = threadIdx.x;
  const int lane = tid & 31;
  const int wv   = tid >> 5;
  const int NT   = (3 * Hc) / 64;               // 96 N-blocks
  const int mb = blockIdx.x / NT, nt = blockIdx.x % NT;
  const int m0 = mb * 128 + wv * 32, n0 = nt * 64;
  const int hw = lane >> 4, l16 = lane & 15;
  const int kc0 = 16 * hw, kc1 = 16 * hw + 8;   // A chunk offsets within 32-K step
  constexpr int NSTEP = Hc / 32;                // 64 K-steps

  f32x8 acc[2][4];
#pragma unroll
  for (int mi = 0; mi < 2; ++mi)
#pragma unroll
    for (int nj = 0; nj < 4; ++nj) acc[mi][nj] = f32x8{0,0,0,0,0,0,0,0};

  const float* arow0 = hidden + (size_t)(m0 + l16) * Hc;
  const float* arow1 = hidden + (size_t)(m0 + 16 + l16) * Hc;

  async_copy_b_tile(w16, n0, 0, Bsh[0], tid);

  for (int i = 0; i < NSTEP; ++i) {
    const int k0  = i * 32;
    const int cur = i & 1;
    const bool has_next = (i + 1 < NSTEP);
    if (has_next) {
      async_copy_b_tile(w16, n0, k0 + 32, Bsh[cur ^ 1], tid);
      wait_async_le2();
    } else {
      wait_async_le0();
    }
    __syncthreads();   // tile i resident in LDS for all waves

    bf16x16 a0 = afrag_f32(arow0, k0 + kc0, k0 + kc1);
    bf16x16 a1 = afrag_f32(arow1, k0 + kc0, k0 + kc1);
    const unsigned short* bb = Bsh[cur];
#pragma unroll
    for (int nj = 0; nj < 4; ++nj) {
      bf16x16 b = bfrag(bb + (16 * nj + l16) * BROW + 16 * hw);
      acc[0][nj] = wmma_bf16(a0, b, acc[0][nj]);
      acc[1][nj] = wmma_bf16(a1, b, acc[1][nj]);
    }
    __syncthreads();   // all waves done reading tile i before it is overwritten
  }

#pragma unroll
  for (int nj = 0; nj < 4; ++nj) {
    const int n = n0 + 16 * nj + l16;
    const int which = n >> 11;          // 0=q, 1=k, 2=v
    const int f = n & (Hc - 1);
    const int h = f >> 6, d = f & 63;
#pragma unroll
    for (int mi = 0; mi < 2; ++mi) {
#pragma unroll
      for (int r = 0; r < 8; ++r) {
        const int m  = m0 + 16 * mi + r + 8 * hw;
        const int bbx = m >> 11, s = m & (Sc - 1);
        const unsigned short v = f2bf_bits(acc[mi][nj][r]);
        if (which == 0)
          Qp[(((size_t)(bbx * NHc + h)) * Sc + s) * HDc + d] = v;
        else if (which == 1)
          Kp[(((size_t)(bbx * NHc + h)) * Sc + s) * HDc + d] = v;
        else
          Vt[(((size_t)(bbx * NHc + h)) * HDc + d) * Sc + s] = v;
      }
    }
  }
}

// ---------------- 3) RoPE on dims [0,16) of Q and K ----------------
__global__ void rope_kernel(unsigned short* __restrict__ Qp,
                            unsigned short* __restrict__ Kp,
                            const int* __restrict__ pos_ids) {
  const int total = Bc * NHc * Sc * 8;
  int idx = blockIdx.x * blockDim.x + threadIdx.x;
  if (idx >= total) return;
  const int d = idx & 7;
  int t = idx >> 3;
  const int s = t & (Sc - 1); t >>= 11;
  const int h = t & (NHc - 1);
  const int b = t >> 5;
  const int pos = pos_ids[b * Sc + s];
  const float inv = __powf(10000.0f, -(float)d * 0.125f);  // theta^(-d/8)
  float c, sn;
  __sincosf((float)pos * inv, &c, &sn);
  const size_t base = (((size_t)(b * NHc + h)) * Sc + s) * HDc;
  {
    float x1 = bf2f(Qp[base + d]), x2 = bf2f(Qp[base + d + 8]);
    Qp[base + d]     = f2bf_bits(x1 * c - x2 * sn);
    Qp[base + d + 8] = f2bf_bits(x2 * c + x1 * sn);
  }
  {
    float x1 = bf2f(Kp[base + d]), x2 = bf2f(Kp[base + d + 8]);
    Kp[base + d]     = f2bf_bits(x1 * c - x2 * sn);
    Kp[base + d + 8] = f2bf_bits(x2 * c + x1 * sn);
  }
}

// ---------------- 4) Flash attention (causal), one wave = 32 queries ----------------
__global__ __launch_bounds__(128) void attn_kernel(
    const unsigned short* __restrict__ Qp, const unsigned short* __restrict__ Kp,
    const unsigned short* __restrict__ Vt, unsigned short* __restrict__ attnout) {
  __shared__ unsigned short Plds[4][32 * 40];   // per-wave 32x32 P tile, padded rows
  const int lane  = threadIdx.x & 31;
  const int wslot = threadIdx.x >> 5;
  const int wid   = blockIdx.x * 4 + wslot;
  const int QB = Sc / 32;                        // 64 query blocks per (b,h)
  const int qb = wid % QB;
  int t = wid / QB;
  const int h = t % NHc, b = t / NHc;
  const int q0 = qb * 32;
  const int hw = lane >> 4, l16 = lane & 15;

  const size_t bh = (size_t)(b * NHc + h);
  const unsigned short* Qb_ = Qp + bh * Sc * HDc;
  const unsigned short* Kb_ = Kp + bh * Sc * HDc;
  const unsigned short* Vb_ = Vt + bh * HDc * Sc;

  // Q fragments: [mi][dstep], reused across all key blocks.
  bf16x16 qf[2][2];
#pragma unroll
  for (int mi = 0; mi < 2; ++mi)
#pragma unroll
    for (int tt = 0; tt < 2; ++tt)
      qf[mi][tt] = afrag_bf16(Qb_ + (size_t)(q0 + 16 * mi + l16) * HDc,
                              32 * tt + 16 * hw, 32 * tt + 16 * hw + 8);

  f32x8 o[2][4];
  f32x8 mrow[2], lrow[2];
#pragma unroll
  for (int mi = 0; mi < 2; ++mi) {
#pragma unroll
    for (int nj = 0; nj < 4; ++nj) o[mi][nj] = f32x8{0,0,0,0,0,0,0,0};
#pragma unroll
    for (int r = 0; r < 8; ++r) { mrow[mi][r] = -1e30f; lrow[mi][r] = 0.0f; }
  }

  unsigned short* pl = Plds[wslot];
  const float scale = 0.125f;  // 1/sqrt(64)

  for (int kb2 = 0; kb2 <= qb; ++kb2) {
    const int k0 = kb2 * 32;
    // ---- scores S = Q * K^T (scaled) ----
    bf16x16 kf[2][2];
#pragma unroll
    for (int nj = 0; nj < 2; ++nj)
#pragma unroll
      for (int tt = 0; tt < 2; ++tt)
        kf[nj][tt] = bfrag(Kb_ + (size_t)(k0 + 16 * nj + l16) * HDc + 32 * tt + 16 * hw);

    f32x8 sc[2][2];
#pragma unroll
    for (int mi = 0; mi < 2; ++mi)
#pragma unroll
      for (int nj = 0; nj < 2; ++nj) {
        f32x8 z = f32x8{0,0,0,0,0,0,0,0};
        z = wmma_bf16(qf[mi][0], kf[nj][0], z);
        sc[mi][nj] = wmma_bf16(qf[mi][1], kf[nj][1], z);
      }

    // ---- scale + causal mask (diagonal block only) ----
#pragma unroll
    for (int mi = 0; mi < 2; ++mi)
#pragma unroll
      for (int nj = 0; nj < 2; ++nj)
#pragma unroll
        for (int r = 0; r < 8; ++r) {
          float v = sc[mi][nj][r] * scale;
          if (kb2 == qb) {
            const int key = k0 + 16 * nj + l16;
            const int qq  = q0 + 16 * mi + r + 8 * hw;
            if (key > qq) v = -1e30f;
          }
          sc[mi][nj][r] = v;
        }

    // ---- online softmax ----
#pragma unroll
    for (int mi = 0; mi < 2; ++mi) {
      f32x8 rm;
#pragma unroll
      for (int r = 0; r < 8; ++r) rm[r] = fmaxf(sc[mi][0][r], sc[mi][1][r]);
#pragma unroll
      for (int off = 1; off < 16; off <<= 1)
#pragma unroll
        for (int r = 0; r < 8; ++r) rm[r] = fmaxf(rm[r], __shfl_xor(rm[r], off, 32));
      f32x8 alpha;
#pragma unroll
      for (int r = 0; r < 8; ++r) {
        const float mn = fmaxf(mrow[mi][r], rm[r]);
        alpha[r] = __expf(mrow[mi][r] - mn);
        mrow[mi][r] = mn;
      }
      f32x8 rs = f32x8{0,0,0,0,0,0,0,0};
#pragma unroll
      for (int nj = 0; nj < 2; ++nj)
#pragma unroll
        for (int r = 0; r < 8; ++r) {
          const float p = __expf(sc[mi][nj][r] - mrow[mi][r]);
          sc[mi][nj][r] = p;
          rs[r] += p;
        }
#pragma unroll
      for (int off = 1; off < 16; off <<= 1)
#pragma unroll
        for (int r = 0; r < 8; ++r) rs[r] += __shfl_xor(rs[r], off, 32);
#pragma unroll
      for (int r = 0; r < 8; ++r) lrow[mi][r] = lrow[mi][r] * alpha[r] + rs[r];
#pragma unroll
      for (int nj = 0; nj < 4; ++nj)
#pragma unroll
        for (int r = 0; r < 8; ++r) o[mi][nj][r] *= alpha[r];
    }

    // ---- P: C-layout -> LDS -> A-layout (wave-private slot, DS fence) ----
#pragma unroll
    for (int mi = 0; mi < 2; ++mi)
#pragma unroll
      for (int nj = 0; nj < 2; ++nj)
#pragma unroll
        for (int r = 0; r < 8; ++r) {
          const int ml = 16 * mi + r + 8 * hw;
          const int nl = 16 * nj + l16;
          pl[ml * 40 + nl] = f2bf_bits(sc[mi][nj][r]);
        }
    asm volatile("s_wait_dscnt 0" ::: "memory");

    bf16x16 pa[2];
#pragma unroll
    for (int mi = 0; mi < 2; ++mi)
      pa[mi] = afrag_bf16(pl + (16 * mi + l16) * 40, 16 * hw, 16 * hw + 8);

    // ---- O += P * V ----
#pragma unroll
    for (int nj2 = 0; nj2 < 4; ++nj2) {
      bf16x16 vb = bfrag(Vb_ + (size_t)(16 * nj2 + l16) * Sc + k0 + 16 * hw);
      o[0][nj2] = wmma_bf16(pa[0], vb, o[0][nj2]);
      o[1][nj2] = wmma_bf16(pa[1], vb, o[1][nj2]);
    }
  }

  // ---- normalize + store bf16 to [b, s, h*64+d] ----
#pragma unroll
  for (int mi = 0; mi < 2; ++mi)
#pragma unroll
    for (int nj = 0; nj < 4; ++nj)
#pragma unroll
      for (int r = 0; r < 8; ++r) {
        const int s  = q0 + 16 * mi + r + 8 * hw;
        const int dd = 16 * nj + l16;
        const float val = o[mi][nj][r] / lrow[mi][r];
        attnout[((size_t)(b * Sc + s)) * Hc + h * HDc + dd] = f2bf_bits(val);
      }
}

// ---------------- 5) Output projection: out = attn @ o_w^T (fp32 out) ----------------
__global__ __launch_bounds__(128) void gemm_out_kernel(
    const unsigned short* __restrict__ A16, const unsigned short* __restrict__ w16,
    float* __restrict__ out) {
  __shared__ unsigned short Bsh[2][64 * BROW];
  const int tid  = threadIdx.x;
  const int lane = tid & 31;
  const int wv   = tid >> 5;
  const int NT   = Hc / 64;               // 32 N-blocks
  const int mb = blockIdx.x / NT, nt = blockIdx.x % NT;
  const int m0 = mb * 128 + wv * 32, n0 = nt * 64;
  const int hw = lane >> 4, l16 = lane & 15;
  const int kc0 = 16 * hw, kc1 = 16 * hw + 8;
  constexpr int NSTEP = Hc / 32;

  f32x8 acc[2][4];
#pragma unroll
  for (int mi = 0; mi < 2; ++mi)
#pragma unroll
    for (int nj = 0; nj < 4; ++nj) acc[mi][nj] = f32x8{0,0,0,0,0,0,0,0};

  const unsigned short* arow0 = A16 + (size_t)(m0 + l16) * Hc;
  const unsigned short* arow1 = A16 + (size_t)(m0 + 16 + l16) * Hc;

  async_copy_b_tile(w16, n0, 0, Bsh[0], tid);

  for (int i = 0; i < NSTEP; ++i) {
    const int k0  = i * 32;
    const int cur = i & 1;
    const bool has_next = (i + 1 < NSTEP);
    if (has_next) {
      async_copy_b_tile(w16, n0, k0 + 32, Bsh[cur ^ 1], tid);
      wait_async_le2();
    } else {
      wait_async_le0();
    }
    __syncthreads();

    bf16x16 a0 = afrag_bf16(arow0, k0 + kc0, k0 + kc1);
    bf16x16 a1 = afrag_bf16(arow1, k0 + kc0, k0 + kc1);
    const unsigned short* bb = Bsh[cur];
#pragma unroll
    for (int nj = 0; nj < 4; ++nj) {
      bf16x16 b = bfrag(bb + (16 * nj + l16) * BROW + 16 * hw);
      acc[0][nj] = wmma_bf16(a0, b, acc[0][nj]);
      acc[1][nj] = wmma_bf16(a1, b, acc[1][nj]);
    }
    __syncthreads();
  }

#pragma unroll
  for (int nj = 0; nj < 4; ++nj) {
    const int n = n0 + 16 * nj + l16;
#pragma unroll
    for (int mi = 0; mi < 2; ++mi)
#pragma unroll
      for (int r = 0; r < 8; ++r) {
        const int m = m0 + 16 * mi + r + 8 * hw;
        out[(size_t)m * Hc + n] = acc[mi][nj][r];
      }
  }
}

// ---------------- host-side launcher ----------------
extern "C" void kernel_launch(void* const* d_in, const int* in_sizes, int n_in,
                              void* d_out, int out_size, void* d_ws, size_t ws_size,
                              hipStream_t stream) {
  const float* hidden = (const float*)d_in[0];
  const float* qkv_w  = (const float*)d_in[1];
  const float* o_w    = (const float*)d_in[2];
  const int*   pos    = (const int*)d_in[3];
  float* out = (float*)d_out;

  unsigned short* ws = (unsigned short*)d_ws;
  const size_t nQKVW = (size_t)3 * Hc * Hc;          // 12.58M
  const size_t nOW   = (size_t)Hc * Hc;              // 4.19M
  const size_t nAct  = (size_t)Bc * NHc * Sc * HDc;  // 8.39M
  unsigned short* qkvw16  = ws;
  unsigned short* ow16    = qkvw16 + nQKVW;
  unsigned short* Qp      = ow16 + nOW;
  unsigned short* Kp      = Qp + nAct;
  unsigned short* Vt      = Kp + nAct;
  unsigned short* attnout = Vt + nAct;

  {
    int n = (int)nQKVW;
    f32_to_bf16_kernel<<<(n + 255) / 256, 256, 0, stream>>>(qkv_w, qkvw16, n);
  }
  {
    int n = (int)nOW;
    f32_to_bf16_kernel<<<(n + 255) / 256, 256, 0, stream>>>(o_w, ow16, n);
  }

  // QKV GEMM: M=4096 (32 macro-blocks of 128), N=6144 (96 n-blocks of 64)
  gemm_qkv_kernel<<<32 * 96, 128, 0, stream>>>(hidden, qkvw16, Qp, Kp, Vt);

  // RoPE
  {
    int n = Bc * NHc * Sc * 8;
    rope_kernel<<<(n + 255) / 256, 256, 0, stream>>>(Qp, Kp, pos);
  }

  // Attention: B*NH*(S/32) = 4096 waves, 4 per block
  attn_kernel<<<4096 / 4, 128, 0, stream>>>(Qp, Kp, Vt, attnout);

  // O-proj GEMM: M=4096 (32 macro-blocks of 128), N=2048 (32 n-blocks of 64)
  gemm_out_kernel<<<32 * 32, 128, 0, stream>>>(attnout, ow16, out);
}
